// Inter_modal_91302414778536
// MI455X (gfx1250) — matrix operand order repaired and checked by reading, not compiled, not run on hardware
//
#include <hip/hip_runtime.h>
#include <hip/hip_bf16.h>
#include <math.h>

typedef __attribute__((ext_vector_type(16))) _Float16 v16h;
typedef __attribute__((ext_vector_type(8)))  float    v8f;

#define TAUF 3.0f
#define NEG_INF (-3.0e38f)

// ---------------------------------------------------------------------------
// Normalize rows over D=1024 (L2, norm clamped at 1e-8) and pack to f16 in
// WMMA fragment layout with contiguous k (k = 16*(lane/16) + j).
//   in : [B][rows][1024] f32
//   out: [B][T][32 ksteps][512 halves]  (fragment element (lane,j) at lane*16+j)
// A-role (isB=0): 16x32 A-matrix map; B-role (isB=1): 32x16 B-matrix map.
// Padded rows (row >= rows) are written as zeros.
// ---------------------------------------------------------------------------
__global__ __launch_bounds__(256) void norm_pack_kernel(
    const float* __restrict__ in, _Float16* __restrict__ out,
    int rows, int T, int isB)
{
  int b   = blockIdx.y;
  int row = blockIdx.x;            // padded row: 0 .. T*16-1
  int tid = threadIdx.x;           // 256 threads, 4 elems each
  __shared__ float red[256];

  bool real = row < rows;
  const float* src = in + ((size_t)b * rows + (real ? row : 0)) * 1024;

  float vals[4];
  float ss = 0.f;
#pragma unroll
  for (int i = 0; i < 4; ++i) {
    float v = real ? src[tid * 4 + i] : 0.f;
    vals[i] = v;
    ss += v * v;
  }
  red[tid] = ss;
  __syncthreads();
  for (int s = 128; s > 0; s >>= 1) {
    if (tid < s) red[tid] += red[tid + s];
    __syncthreads();
  }
  float inv = 1.0f / fmaxf(sqrtf(red[0]), 1e-8f);

  int rt = row >> 4, m = row & 15;
  size_t tile = ((size_t)b * T + rt) * (32 * 512);
#pragma unroll
  for (int i = 0; i < 4; ++i) {
    int d = tid * 4 + i;
    int kstep = d >> 5, kk = d & 31;
    int lane, j;
    if (isB) {                       // B 32x16: lane<16 -> K=0..15, lane>=16 -> K=16..31
      lane = m + ((kk >= 16) ? 16 : 0);
      j = kk & 15;
    } else {                         // A 16x32: lane<16 -> K=0..15, lane>=16 -> K=16..31
      lane = m + ((kk >= 16) ? 16 : 0);
      j = kk & 15;
    }
    out[tile + (size_t)kstep * 512 + lane * 16 + j] = (_Float16)(vals[i] * inv);
  }
}

// ---------------------------------------------------------------------------
// Stage-1 all-pairs scorer with 2x2 (n x p) register blocking.
//   out[n*P + p] = sum_l max_r dot(A_n[r], B_p[l])
// Block = 8 waves; block covers (n0, n0+1) x 16 consecutive p.  Both A r-tiles
// staged in 64 KB LDS; each global B fragment feeds 2 WMMAs, each LDS A
// fragment feeds 2 WMMAs -> 1 global b128-pair + 1 ds b128-pair per 2 WMMAs.
// Requires P % 16 == 0 and N % 2 == 0 (true for stage 1: P = N = 256).
// ---------------------------------------------------------------------------
__global__ __launch_bounds__(256) void sim2x2_kernel(
    const _Float16* __restrict__ Apk, const _Float16* __restrict__ Bpk,
    float* __restrict__ out, int P, int RT, int LT, int realR)
{
  __shared__ __align__(32) _Float16 lds[2 * 32 * 512];   // two A r-tiles: 64 KB

  int lane = threadIdx.x & 31;
  int wave = threadIdx.x >> 5;
  int n0   = blockIdx.y * 2;
  int p0   = blockIdx.x * 16 + wave * 2;   // this wave: p0 and p0+1

  const v16h* Bv0 = (const v16h*)(Bpk + (size_t)(p0 + 0) * LT * (32 * 512));
  const v16h* Bv1 = (const v16h*)(Bpk + (size_t)(p0 + 1) * LT * (32 * 512));
  const v16h* Av0 = (const v16h*)lds;
  const v16h* Av1 = (const v16h*)lds + 1024;

  float runmax[3][4];
#pragma unroll
  for (int i = 0; i < 3; ++i)
#pragma unroll
    for (int q = 0; q < 4; ++q) runmax[i][q] = NEG_INF;

  int mBase = (lane >= 16) ? 8 : 0;

  for (int rt = 0; rt < RT; ++rt) {
    __syncthreads();
    {   // cooperative load of both A tiles: 2048 v16h (64 KB), 256 threads
      const v16h* s0 = (const v16h*)(Apk + ((size_t)(n0 + 0) * RT + rt) * (32 * 512));
      const v16h* s1 = (const v16h*)(Apk + ((size_t)(n0 + 1) * RT + rt) * (32 * 512));
      v16h* d0 = (v16h*)lds;
      for (int i = threadIdx.x; i < 1024; i += 256) {
        d0[i]        = s0[i];
        d0[i + 1024] = s1[i];
      }
    }
    __syncthreads();

    int validRows = realR - rt * 16;
    if (validRows > 16) validRows = 16;

#pragma unroll
    for (int lt = 0; lt < 3; ++lt) {
      if (lt < LT) {
        v8f acc00 = {}, acc01 = {}, acc10 = {}, acc11 = {};
        const v16h* b0f = Bv0 + (size_t)(lt * 32) * 32 + lane;
        const v16h* b1f = Bv1 + (size_t)(lt * 32) * 32 + lane;
        for (int k = 0; k < 32; ++k) {
          v16h b0 = b0f[(size_t)k * 32];
          v16h b1 = b1f[(size_t)k * 32];
          v16h a0 = Av0[k * 32 + lane];
          v16h a1 = Av1[k * 32 + lane];
          acc00 = __builtin_amdgcn_wmma_f32_16x16x32_f16(false, a0, false, b0, (short)0, acc00, false, false);
          acc01 = __builtin_amdgcn_wmma_f32_16x16x32_f16(false, a0, false, b1, (short)0, acc01, false, false);
          acc10 = __builtin_amdgcn_wmma_f32_16x16x32_f16(false, a1, false, b0, (short)0, acc10, false, false);
          acc11 = __builtin_amdgcn_wmma_f32_16x16x32_f16(false, a1, false, b1, (short)0, acc11, false, false);
        }
        float m00 = NEG_INF, m01 = NEG_INF, m10 = NEG_INF, m11 = NEG_INF;
#pragma unroll
        for (int g = 0; g < 8; ++g) {
          bool ok = (mBase + g) < validRows;     // mask zero-padded A rows
          m00 = ok ? fmaxf(m00, acc00[g]) : m00;
          m01 = ok ? fmaxf(m01, acc01[g]) : m01;
          m10 = ok ? fmaxf(m10, acc10[g]) : m10;
          m11 = ok ? fmaxf(m11, acc11[g]) : m11;
        }
        runmax[lt][0] = fmaxf(runmax[lt][0], m00);
        runmax[lt][1] = fmaxf(runmax[lt][1], m01);
        runmax[lt][2] = fmaxf(runmax[lt][2], m10);
        runmax[lt][3] = fmaxf(runmax[lt][3], m11);
      }
    }
  }

  // combine lane pairs (M 0-7 vs 8-15), then sum the 16 columns per l-tile
  float total[4] = {0.f, 0.f, 0.f, 0.f};
#pragma unroll
  for (int lt = 0; lt < 3; ++lt) {
    if (lt < LT) {
#pragma unroll
      for (int q = 0; q < 4; ++q) {
        float v = runmax[lt][q];
        v = fmaxf(v, __shfl_xor(v, 16, 32));     // colmax duplicated in lane pairs
        for (int off = 1; off < 32; off <<= 1) v += __shfl_xor(v, off, 32);
        total[q] += 0.5f * v;                    // undo lane-pair duplication
      }
    }
  }
  if (lane == 0) {
    out[(size_t)(n0 + 0) * P + (p0 + 0)] = total[0];
    out[(size_t)(n0 + 0) * P + (p0 + 1)] = total[1];
    out[(size_t)(n0 + 1) * P + (p0 + 0)] = total[2];
    out[(size_t)(n0 + 1) * P + (p0 + 1)] = total[3];
  }
}

// ---------------------------------------------------------------------------
// Stage-2 scorer (per-n index map, P=3): out[n*P+pi] with p = pmap[n*P+pi].
// One block per n, 8 waves; inactive waves compute a clamped p harmlessly
// (no early return -> barriers stay uniform, EXEC all-ones around WMMA).
// ---------------------------------------------------------------------------
__global__ __launch_bounds__(256) void sim_kernel(
    const _Float16* __restrict__ Apk, const _Float16* __restrict__ Bpk,
    const int* __restrict__ pmap, float* __restrict__ out,
    int P, int RT, int LT, int realR)
{
  __shared__ __align__(32) _Float16 lds[32 * 512];   // one A r-tile: 32 KB

  int n    = blockIdx.y;
  int lane = threadIdx.x & 31;
  int wave = threadIdx.x >> 5;
  int pi   = blockIdx.x * 8 + wave;
  bool active = pi < P;
  int pcl = active ? pi : (P - 1);
  int p   = pmap ? pmap[(size_t)n * P + pcl] : pcl;

  const v16h* Bv = (const v16h*)(Bpk + (size_t)p * LT * (32 * 512));
  const v16h* Av = (const v16h*)lds;

  float runmax[3];
#pragma unroll
  for (int i = 0; i < 3; ++i) runmax[i] = NEG_INF;

  int mBase = (lane >= 16) ? 8 : 0;

  for (int rt = 0; rt < RT; ++rt) {
    __syncthreads();
    {
      const v16h* s4 = (const v16h*)(Apk + ((size_t)n * RT + rt) * (32 * 512));
      v16h* d4 = (v16h*)lds;
      for (int i = threadIdx.x; i < 1024; i += 256) d4[i] = s4[i];
    }
    __syncthreads();

    int validRows = realR - rt * 16;
    if (validRows > 16) validRows = 16;

#pragma unroll
    for (int lt = 0; lt < 3; ++lt) {
      if (lt < LT) {
        v8f acc = {};
        const v16h* bfrag = Bv + (size_t)(lt * 32) * 32 + lane;
        for (int k = 0; k < 32; ++k) {
          v16h a = Av[k * 32 + lane];
          v16h b = bfrag[(size_t)k * 32];
          acc = __builtin_amdgcn_wmma_f32_16x16x32_f16(
              false, a, false, b, (short)0, acc, false, false);
        }
        float mx = NEG_INF;
#pragma unroll
        for (int g = 0; g < 8; ++g) {
          float v = acc[g];
          mx = (mBase + g < validRows) ? fmaxf(mx, v) : mx;
        }
        runmax[lt] = fmaxf(runmax[lt], mx);
      }
    }
  }

  float total = 0.f;
#pragma unroll
  for (int lt = 0; lt < 3; ++lt) {
    if (lt < LT) {
      float v = runmax[lt];
      v = fmaxf(v, __shfl_xor(v, 16, 32));
      for (int off = 1; off < 32; off <<= 1) v += __shfl_xor(v, off, 32);
      total += 0.5f * v;
    }
  }
  if (active && lane == 0) out[(size_t)n * P + pi] = total;
}

// ---------------------------------------------------------------------------
// Top-3 per row (strict '>' scan -> lowest index wins ties, like lax.top_k).
// ---------------------------------------------------------------------------
__global__ void topk3_kernel(const float* __restrict__ sim,
                             float* __restrict__ vals, int* __restrict__ idx,
                             int N, int P)
{
  int n = blockIdx.x * blockDim.x + threadIdx.x;
  if (n >= N) return;
  const float* row = sim + (size_t)n * P;
  float v0 = NEG_INF, v1 = NEG_INF, v2 = NEG_INF;
  int i0 = 0, i1 = 0, i2 = 0;
  for (int p = 0; p < P; ++p) {
    float x = row[p];
    if (x > v0)      { v2 = v1; i2 = i1; v1 = v0; i1 = i0; v0 = x; i0 = p; }
    else if (x > v1) { v2 = v1; i2 = i1; v1 = x; i1 = p; }
    else if (x > v2) { v2 = x; i2 = p; }
  }
  vals[n * 3 + 0] = v0; vals[n * 3 + 1] = v1; vals[n * 3 + 2] = v2;
  idx [n * 3 + 0] = i0; idx [n * 3 + 1] = i1; idx [n * 3 + 2] = i2;
}

// ---------------------------------------------------------------------------
// KL(p || q) with p=softmax(m1/tau), log_q=log_softmax(m2/tau), summed over
// both directions and all n.  512 threads = 2 dirs x 256 rows, tree-reduced.
// ---------------------------------------------------------------------------
__global__ __launch_bounds__(512) void kl_kernel(
    const float* __restrict__ m1a, const float* __restrict__ m2a,
    const float* __restrict__ m1b, const float* __restrict__ m2b,
    float* __restrict__ out)
{
  __shared__ float red[512];
  int t = threadIdx.x;
  int dir = t >> 8, n = t & 255;
  const float* m1 = dir ? m1b : m1a;
  const float* m2 = dir ? m2b : m2a;

  float a0 = m1[n*3+0] / TAUF, a1 = m1[n*3+1] / TAUF, a2 = m1[n*3+2] / TAUF;
  float b0 = m2[n*3+0] / TAUF, b1 = m2[n*3+1] / TAUF, b2 = m2[n*3+2] / TAUF;

  float am = fmaxf(a0, fmaxf(a1, a2));
  float lseA = am + logf(expf(a0 - am) + expf(a1 - am) + expf(a2 - am));
  float bm = fmaxf(b0, fmaxf(b1, b2));
  float lseB = bm + logf(expf(b0 - bm) + expf(b1 - bm) + expf(b2 - bm));

  float lp0 = a0 - lseA, lp1 = a1 - lseA, lp2 = a2 - lseA;
  float lq0 = b0 - lseB, lq1 = b1 - lseB, lq2 = b2 - lseB;
  float c = expf(lp0) * (lp0 - lq0) + expf(lp1) * (lp1 - lq1) + expf(lp2) * (lp2 - lq2);

  red[t] = c;
  __syncthreads();
  for (int s = 256; s > 0; s >>= 1) {
    if (t < s) red[t] += red[t + s];
    __syncthreads();
  }
  if (t == 0) out[0] = red[0];
}

// ---------------------------------------------------------------------------
extern "C" void kernel_launch(void* const* d_in, const int* in_sizes, int n_in,
                              void* d_out, int out_size, void* d_ws, size_t ws_size,
                              hipStream_t stream)
{
  (void)in_sizes; (void)n_in; (void)out_size; (void)ws_size;

  const float* img_n = (const float*)d_in[0];   // [256,36,1024]
  const float* cap_n = (const float*)d_in[1];   // [256,32,1024]
  const float* img_p = (const float*)d_in[2];   // [256,36,1024]
  const float* cap_p = (const float*)d_in[3];   // [256,32,1024]

  char* ws = (char*)d_ws;
  size_t off = 0;
  auto alloc = [&](size_t bytes) -> void* {
    void* p = ws + off;
    off += (bytes + 255) & ~(size_t)255;
    return p;
  };

  const size_t TILE_H = 32 * 512;                          // halves per 16-row tile
  _Float16* Aimg = (_Float16*)alloc((size_t)256 * 3 * TILE_H * 2);
  _Float16* Acap = (_Float16*)alloc((size_t)256 * 2 * TILE_H * 2);
  _Float16* Bimg = (_Float16*)alloc((size_t)256 * 3 * TILE_H * 2);
  _Float16* Bcap = (_Float16*)alloc((size_t)256 * 2 * TILE_H * 2);
  float* sim1 = (float*)alloc((size_t)256 * 256 * 4);
  float* sim2 = (float*)alloc((size_t)256 * 256 * 4);
  float* tk1v = (float*)alloc(256 * 3 * 4);
  int*   tk1i = (int*)  alloc(256 * 3 * 4);
  float* tk2v = (float*)alloc(256 * 3 * 4);
  int*   tk2i = (int*)  alloc(256 * 3 * 4);
  float* m2k1 = (float*)alloc(256 * 3 * 4);
  float* m2k2 = (float*)alloc(256 * 3 * 4);

  // 1) normalize + pack (A-layout for query side, B-layout for key side)
  norm_pack_kernel<<<dim3(48, 256), 256, 0, stream>>>(img_n, Aimg, 36, 3, 0);
  norm_pack_kernel<<<dim3(32, 256), 256, 0, stream>>>(cap_n, Acap, 32, 2, 0);
  norm_pack_kernel<<<dim3(48, 256), 256, 0, stream>>>(img_p, Bimg, 36, 3, 1);
  norm_pack_kernel<<<dim3(32, 256), 256, 0, stream>>>(cap_p, Bcap, 32, 2, 1);

  // 2) stage-1 all-pairs similarity, 2x2 register-blocked WMMA
  //    grid = (P/16, N/2)
  sim2x2_kernel<<<dim3(16, 128), 256, 0, stream>>>(Aimg, Bimg, sim1, 256, 3, 3, 36);
  sim2x2_kernel<<<dim3(16, 128), 256, 0, stream>>>(Acap, Bcap, sim2, 256, 2, 2, 32);

  // 3) top-3 partners per row
  topk3_kernel<<<1, 256, 0, stream>>>(sim1, tk1v, tk1i, 256, 256);
  topk3_kernel<<<1, 256, 0, stream>>>(sim2, tk2v, tk2i, 256, 256);

  // 4) stage-2: re-score top-3 against the other modality (index-mapped)
  sim_kernel<<<dim3(1, 256), 256, 0, stream>>>(Aimg, Bcap, tk1i, m2k1, 3, 3, 2, 36);
  sim_kernel<<<dim3(1, 256), 256, 0, stream>>>(Acap, Bimg, tk2i, m2k2, 3, 2, 3, 32);

  // 5) KL over both directions -> scalar
  kl_kernel<<<1, 512, 0, stream>>>(tk1v, m2k1, tk2v, m2k2, (float*)d_out);
}
